// RiskPredictionModel_55722905698325
// MI455X (gfx1250) — compile-verified, matrix operand order checked by reading, not run on hardware
//
#include <hip/hip_runtime.h>
#include <hip/hip_bf16.h>

typedef __attribute__((ext_vector_type(16))) _Float16 v16h;
typedef __attribute__((ext_vector_type(8)))  _Float16 v8h;
typedef __attribute__((ext_vector_type(8)))  float    v8f;

#define HID 128
#define NHEAD 4

// ---------------------------------------------------------------------------
// WMMA fragment layouts (CDNA5 16x16x32 f16, ISA 7.12.2)
// A (16x32): lane L -> row (L&15); elements i<8 : K = k0 + 8g + i
//                                   elements i>=8: K = k0 + 16 + 8g + (i-8)
//            (two contiguous 8-float runs -> 4x global_load_b128)
// B (32x16): staged in LDS pre-converted to f16 in fragment order:
//            sW[((ks*8 + nt)*32 + lane)*16 + i]  -> one 32B LDS read per frag
// C/D (16x16 f32): lane L -> col (L&15); VGPR j -> row j + 8*(L>>4)
// ---------------------------------------------------------------------------

template <int PRE_ELU>
__device__ __forceinline__ _Float16 cvt_in(float x) {
    if (PRE_ELU) x = (x > 0.f) ? x : (__expf(x) - 1.f);
    return (_Float16)x;
}

template <int PRE_ELU>
__device__ __forceinline__ v16h load_a_global(const float* X, int ldx, int row0,
                                              int k0, int lane, int nrows) {
    int r = row0 + (lane & 15);
    if (r > nrows - 1) r = nrows - 1;           // clamp tail (stores are guarded)
    const float* rp = X + (size_t)r * ldx + k0 + 8 * ((lane >> 4) & 1);
    float4 f0 = *(const float4*)(rp);
    float4 f1 = *(const float4*)(rp + 4);
    float4 f2 = *(const float4*)(rp + 16);
    float4 f3 = *(const float4*)(rp + 20);
    v16h a;
    a[0]  = cvt_in<PRE_ELU>(f0.x); a[1]  = cvt_in<PRE_ELU>(f0.y);
    a[2]  = cvt_in<PRE_ELU>(f0.z); a[3]  = cvt_in<PRE_ELU>(f0.w);
    a[4]  = cvt_in<PRE_ELU>(f1.x); a[5]  = cvt_in<PRE_ELU>(f1.y);
    a[6]  = cvt_in<PRE_ELU>(f1.z); a[7]  = cvt_in<PRE_ELU>(f1.w);
    a[8]  = cvt_in<PRE_ELU>(f2.x); a[9]  = cvt_in<PRE_ELU>(f2.y);
    a[10] = cvt_in<PRE_ELU>(f2.z); a[11] = cvt_in<PRE_ELU>(f2.w);
    a[12] = cvt_in<PRE_ELU>(f3.x); a[13] = cvt_in<PRE_ELU>(f3.y);
    a[14] = cvt_in<PRE_ELU>(f3.z); a[15] = cvt_in<PRE_ELU>(f3.w);
    return a;
}

// A-fragment from an LDS-resident 16x128 f16 tile (row-major)
__device__ __forceinline__ v16h load_a_lds(const _Float16* sh, int k0, int lane) {
    int r = lane & 15;
    int g = (lane >> 4) & 1;
    const _Float16* p = sh + r * HID + k0 + 8 * g;
    v8h lo = *(const v8h*)(p);
    v8h hi = *(const v8h*)(p + 16);
    v16h a;
#pragma unroll
    for (int i = 0; i < 8; ++i) { a[i] = lo[i]; a[8 + i] = hi[i]; }
    return a;
}

// Cooperative: global W[K x 128] (f32) -> LDS f16 in B-fragment order.
// Coalesced float4 reads along N, scattered b16 LDS writes. Once per block.
__device__ __forceinline__ void stage_w_lds(_Float16* sW, const float* __restrict__ W,
                                            int K, int tid, int nthreads) {
    int total = K * HID;
    for (int idx = tid * 4; idx < total; idx += nthreads * 4) {
        int k  = idx >> 7;           // row of W
        int n0 = idx & 127;          // col (multiple of 4)
        float4 w = *(const float4*)(W + (size_t)k * HID + n0);
        int ks = k >> 5, kk = k & 31;
        int g, i;
        if (kk < 16) { g = kk >> 3; i = kk & 7; }
        else { int k2 = kk - 16; g = k2 >> 3; i = 8 + (k2 & 7); }
#pragma unroll
        for (int u = 0; u < 4; ++u) {
            int n = n0 + u;
            float wv = (u == 0) ? w.x : (u == 1) ? w.y : (u == 2) ? w.z : w.w;
            int lane = (g << 4) | (n & 15);
            int nt = n >> 4;
            sW[(((ks * 8 + nt) * 32) + lane) * 16 + i] = (_Float16)wv;
        }
    }
}

__device__ __forceinline__ v16h load_b_frag_lds(const _Float16* sW, int ks, int nt, int lane) {
    return *(const v16h*)(sW + (((ks * 8 + nt) * 32) + lane) * 16);
}

// ---------------------------------------------------------------------------
// Generic Y[N x 128] = post( pre(X[N x K]) @ W[K x 128] + bias )
// W staged in LDS (f16, fragment order). One wave -> 16 rows x 128 cols.
// ---------------------------------------------------------------------------
template <int PRE_ELU, int POST_RELU>
__global__ void wmma_gemm128(const float* __restrict__ X, const float* __restrict__ W,
                             const float* __restrict__ bias, float* __restrict__ Y,
                             int N, int K) {
    __shared__ _Float16 sW[128 * HID];          // 32 KB (K <= 128)
    int tid = threadIdx.x;
    stage_w_lds(sW, W, K, tid, blockDim.x);
    __syncthreads();

    int lane = tid & 31;
    int wave = tid >> 5;
    int row0 = (blockIdx.x * (blockDim.x >> 5) + wave) * 16;
    if (row0 >= N) return;

    v8f acc[8];
#pragma unroll
    for (int t = 0; t < 8; ++t) acc[t] = (v8f){0.f,0.f,0.f,0.f,0.f,0.f,0.f,0.f};

    for (int k0 = 0; k0 < K; k0 += 32) {
        if (k0 + 32 < K)
            __builtin_prefetch(X + (size_t)(row0 + (lane & 15)) * K + k0 + 32, 0, 3);
        v16h a = load_a_global<PRE_ELU>(X, K, row0, k0, lane, N);
        int ks = k0 >> 5;
#pragma unroll
        for (int nt = 0; nt < 8; ++nt) {
            v16h b = load_b_frag_lds(sW, ks, nt, lane);
            acc[nt] = __builtin_amdgcn_wmma_f32_16x16x32_f16(
                false, a, false, b, (short)0, acc[nt], false, false);
        }
    }

    int n = lane & 15, g = (lane >> 4) & 1;
#pragma unroll
    for (int nt = 0; nt < 8; ++nt) {
        int col = nt * 16 + n;
        float bv = bias ? bias[col] : 0.f;
#pragma unroll
        for (int j = 0; j < 8; ++j) {
            int row = row0 + j + 8 * g;
            if (row < N) {
                float v = acc[nt][j] + bv;
                if (POST_RELU) v = fmaxf(v, 0.f);
                Y[(size_t)row * HID + col] = v;
            }
        }
    }
}

// ---------------------------------------------------------------------------
// Fused: S[N x 128] = relu(elu(X[N x 32]) @ Wp[32 x 128] + bp) @ W2a[128 x 128]
// h never touches HBM: f32 tile -> f16 LDS tile -> A-operand of 2nd GEMM.
// Block = 128 threads = 4 waves. LDS: 32 KB staged W2a + 16 KB region that
// first holds staged Wp (8 KB, stage-1 B operand), then the 4 h tiles.
// ---------------------------------------------------------------------------
__global__ void fused_proj_tecsrc(const float* __restrict__ X, const float* __restrict__ Wp,
                                  const float* __restrict__ bp, const float* __restrict__ W2a,
                                  float* __restrict__ S, int N) {
    __shared__ _Float16 sW2[128 * HID];         // 32 KB, fragment order
    __shared__ _Float16 sh[4 * 16 * HID];       // 16 KB: Wp staging, then h tiles
    int tid = threadIdx.x;
    stage_w_lds(sW2, W2a, 128, tid, blockDim.x);
    stage_w_lds(sh, Wp, 32, tid, blockDim.x);   // Wp -> first 8 KB of sh
    __syncthreads();

    int lane = tid & 31;
    int wave = tid >> 5;
    int row0 = (blockIdx.x * 4 + wave) * 16;
    _Float16* myh = sh + wave * 16 * HID;

    // ---- stage 1: h = relu(elu(X) @ Wp + bp), single K=32 step
    v8f acc[8];
#pragma unroll
    for (int t = 0; t < 8; ++t) acc[t] = (v8f){0.f,0.f,0.f,0.f,0.f,0.f,0.f,0.f};
    {
        v16h a = load_a_global<1>(X, 32, row0, 0, lane, N);
#pragma unroll
        for (int nt = 0; nt < 8; ++nt) {
            v16h b = load_b_frag_lds(sh, 0, nt, lane);
            acc[nt] = __builtin_amdgcn_wmma_f32_16x16x32_f16(
                false, a, false, b, (short)0, acc[nt], false, false);
        }
    }
    __syncthreads();                            // all waves done reading Wp frags

    int n = lane & 15, g = (lane >> 4) & 1;
#pragma unroll
    for (int nt = 0; nt < 8; ++nt) {
        int col = nt * 16 + n;
        float bv = bp[col];
#pragma unroll
        for (int j = 0; j < 8; ++j) {
            float v = fmaxf(acc[nt][j] + bv, 0.f);
            myh[(j + 8 * g) * HID + col] = (_Float16)v;
        }
    }
    __syncthreads();                            // h tiles visible (and W2a staged)

    // ---- stage 2: S = h @ W2a, K = 128
    v8f acc2[8];
#pragma unroll
    for (int t = 0; t < 8; ++t) acc2[t] = (v8f){0.f,0.f,0.f,0.f,0.f,0.f,0.f,0.f};
#pragma unroll
    for (int k0 = 0; k0 < HID; k0 += 32) {
        v16h a = load_a_lds(myh, k0, lane);
#pragma unroll
        for (int nt = 0; nt < 8; ++nt) {
            v16h b = load_b_frag_lds(sW2, k0 >> 5, nt, lane);
            acc2[nt] = __builtin_amdgcn_wmma_f32_16x16x32_f16(
                false, a, false, b, (short)0, acc2[nt], false, false);
        }
    }
#pragma unroll
    for (int nt = 0; nt < 8; ++nt) {
        int col = nt * 16 + n;
#pragma unroll
        for (int j = 0; j < 8; ++j) {
            int row = row0 + j + 8 * g;
            if (row < N) S[(size_t)row * HID + col] = acc2[nt][j];
        }
    }
}

// ---------------------------------------------------------------------------
// TEC helpers
// ---------------------------------------------------------------------------
__global__ void tec_fuse_w(const float* __restrict__ W1, const float* __restrict__ b1,
                           const float* __restrict__ W2, const float* __restrict__ b2,
                           float* __restrict__ Wc, float* __restrict__ bc, int edim) {
    int j = threadIdx.x;                       // 128 threads
    const float* W2e = W2 + 256 * HID;         // rows [256, 768): edge-feature block
    float ab = b2[j];
    for (int k = 0; k < 512; ++k) ab += b1[k] * W2e[(size_t)k * HID + j];
    bc[j] = ab;
    for (int r = 0; r < edim; ++r) {
        float a = 0.f;
        for (int k = 0; k < 512; ++k) a += W1[(size_t)r * 512 + k] * W2e[(size_t)k * HID + j];
        Wc[r * HID + j] = a;
    }
}

__global__ void tec_scatter(const int* __restrict__ src, const int* __restrict__ dst,
                            const float* __restrict__ ea, int edim,
                            const float* __restrict__ S, const float* __restrict__ D,
                            const float* __restrict__ Wc, const float* __restrict__ bc,
                            float* __restrict__ hacc, int E) {
    int t = blockIdx.x * blockDim.x + threadIdx.x;
    int e = t >> 5;
    if (e >= E) return;
    int c0 = (t & 31) * 4;
    int s = src[e], d = dst[e];
    float eav[6];
    for (int r = 0; r < edim; ++r) eav[r] = ea[(size_t)e * edim + r];
    float4 sv = *(const float4*)(S + (size_t)s * HID + c0);
    float4 dv = *(const float4*)(D + (size_t)d * HID + c0);
    float vs[4] = {sv.x + dv.x, sv.y + dv.y, sv.z + dv.z, sv.w + dv.w};
#pragma unroll
    for (int u = 0; u < 4; ++u) {
        int c = c0 + u;
        float v = vs[u] + bc[c];
        for (int r = 0; r < edim; ++r) v += eav[r] * Wc[r * HID + c];
        atomicAdd(&hacc[(size_t)d * HID + c], v);
    }
}

// ---------------------------------------------------------------------------
// GAT helpers
// ---------------------------------------------------------------------------
__device__ __forceinline__ void atomicMaxFloat(float* addr, float val) {
    unsigned* u = (unsigned*)addr;
    unsigned old = *u;
    while (__uint_as_float(old) < val) {
        unsigned assumed = old;
        old = atomicCAS(u, assumed, __float_as_uint(val));
        if (old == assumed) break;
    }
}

__global__ void fill_f32(float* __restrict__ p, float v, int n) {
    int t = blockIdx.x * blockDim.x + threadIdx.x;
    if (t < n) p[t] = v;
}

__global__ void attn_score(const float* __restrict__ Z, const float* __restrict__ av,
                           float* __restrict__ out, int N) {
    int t = blockIdx.x * blockDim.x + threadIdx.x;
    if (t >= N * NHEAD) return;
    int nidx = t >> 2, h = t & 3;
    const float* z = Z + (size_t)nidx * HID + h * 32;
    const float* a = av + h * 32;
    float s = 0.f;
#pragma unroll
    for (int d = 0; d < 32; ++d) s += z[d] * a[d];
    out[t] = s;
}

__global__ void gat_init(float* __restrict__ gout, const float* __restrict__ hdst,
                         const float* __restrict__ b, int N) {
    int t = blockIdx.x * blockDim.x + threadIdx.x;
    if (t >= N * HID) return;
    gout[t] = hdst[t] + b[t & (HID - 1)];
}

__global__ void gat_edge_max(const int* __restrict__ src, const int* __restrict__ dst,
                             const float* __restrict__ el, const float* __restrict__ er,
                             float* __restrict__ score, float* __restrict__ emax, int E) {
    int t = blockIdx.x * blockDim.x + threadIdx.x;
    if (t >= E * NHEAD) return;
    int e = t >> 2, h = t & 3;
    float s = el[src[e] * NHEAD + h] + er[dst[e] * NHEAD + h];
    s = (s > 0.f) ? s : 0.2f * s;               // leaky_relu(0.2)
    score[t] = s;
    atomicMaxFloat(&emax[dst[e] * NHEAD + h], s);
}

__global__ void gat_edge_exp(const int* __restrict__ dst, float* __restrict__ score,
                             const float* __restrict__ emax, float* __restrict__ denom, int E) {
    int t = blockIdx.x * blockDim.x + threadIdx.x;
    if (t >= E * NHEAD) return;
    int e = t >> 2, h = t & 3;
    float v = __expf(score[t] - emax[dst[e] * NHEAD + h]);
    score[t] = v;
    atomicAdd(&denom[dst[e] * NHEAD + h], v);
}

__global__ void gat_edge_scatter(const int* __restrict__ src, const int* __restrict__ dst,
                                 const float* __restrict__ score, const float* __restrict__ denom,
                                 const float* __restrict__ Zs, float* __restrict__ gout, int E) {
    int t = blockIdx.x * blockDim.x + threadIdx.x;
    int e = t >> 5;
    if (e >= E) return;
    int c0 = (t & 31) * 4;
    int s = src[e], d = dst[e];
    float4 zv = *(const float4*)(Zs + (size_t)s * HID + c0);
    float zs4[4] = {zv.x, zv.y, zv.z, zv.w};
#pragma unroll
    for (int u = 0; u < 4; ++u) {
        int c = c0 + u, h = c >> 5;
        float alpha = score[e * NHEAD + h] / denom[d * NHEAD + h];
        atomicAdd(&gout[(size_t)d * HID + c], alpha * zs4[u]);
    }
}

// ---------------------------------------------------------------------------
// Readout: mean over company nodes -> linear(128,10) -> log_softmax
// ---------------------------------------------------------------------------
__global__ void readout(const float* __restrict__ G, int Nc, const float* __restrict__ Wf,
                        const float* __restrict__ bf, float* __restrict__ out) {
    __shared__ float agg[HID];
    __shared__ float logits[10];
    int c = threadIdx.x;                        // 128 threads
    float s = 0.f;
    for (int i = 0; i < Nc; ++i) s += G[(size_t)i * HID + c];
    agg[c] = s / (float)Nc;
    __syncthreads();
    if (c < 10) {
        float l = bf[c];
        for (int k = 0; k < HID; ++k) l += agg[k] * Wf[k * 10 + c];
        logits[c] = l;
    }
    __syncthreads();
    if (c == 0) {
        float m = -3.0e38f;
        for (int j = 0; j < 10; ++j) m = fmaxf(m, logits[j]);
        float se = 0.f;
        for (int j = 0; j < 10; ++j) se += __expf(logits[j] - m);
        float lse = m + __logf(se);
        for (int j = 0; j < 10; ++j) out[j] = logits[j] - lse;
    }
}

// ---------------------------------------------------------------------------
// Host orchestration
// ---------------------------------------------------------------------------
extern "C" void kernel_launch(void* const* d_in, const int* in_sizes, int n_in,
                              void* d_out, int out_size, void* d_ws, size_t ws_size,
                              hipStream_t stream) {
    auto F = [&](int i) { return (const float*)d_in[i]; };
    auto I = [&](int i) { return (const int*)d_in[i]; };

    // Input layout (setup_inputs insertion order; params nested in insertion order)
    const int Nd  = in_sizes[0] / 32, Nm = in_sizes[1] / 32, Nq = in_sizes[2] / 32;
    const int Ny  = in_sizes[3] / 32, Ncn = in_sizes[4] / 32;
    const int Edm = in_sizes[5], Emq = in_sizes[8], Eqy = in_sizes[11], Eyc = in_sizes[14];
    // 17..26: p1_{day,month,quarter,year,company}{W,b}
    // 27..34: p2_{month,quarter,year,company}{W,b}
    // 35..50: tec_{day,month,quarter,year}{W1,b1,W2,b2}
    // 51..66: gat1_{dm,mq,qy,yc}{W,al,ar,b}; 67..78: gat2_{mq,qy,yc}; 79 Wf, 80 bf
    const int P1 = 17, P2 = 27, TEC = 35, G1 = 51, G2 = 67;

    // Workspace bump allocator
    char* ws = (char*)d_ws;
    size_t off = 0;
    auto alloc = [&](size_t bytes) -> float* {
        float* p = (float*)(ws + off);
        off = (off + bytes + 255) & ~(size_t)255;
        return p;
    };
    float* S_day   = alloc((size_t)Nd * HID * 4);   // only big buffer (~77 MB)
    float* h_month = alloc((size_t)Nm * HID * 4);
    float* h_quart = alloc((size_t)Nq * HID * 4);
    float* h_year  = alloc((size_t)Ny * HID * 4);
    float* h_comp  = alloc((size_t)Ncn * HID * 4);
    float* tmpA    = alloc((size_t)Nm * HID * 4);
    float* tmpB    = alloc((size_t)Nm * HID * 4);
    float* Wc      = alloc(8 * HID * 4);
    float* bc      = alloc(HID * 4);
    float* el      = alloc((size_t)Nm * NHEAD * 4);
    float* er      = alloc((size_t)Nm * NHEAD * 4);
    float* emax    = alloc((size_t)Nm * NHEAD * 4);
    float* denom   = alloc((size_t)Nm * NHEAD * 4);
    float* score   = alloc((size_t)Edm * NHEAD * 4);
    float* gbuf_y  = alloc((size_t)Ny * HID * 4);
    float* gbuf_c  = alloc((size_t)Ncn * HID * 4);
    float* g_year  = alloc((size_t)Ny * HID * 4);
    float* g_comp  = alloc((size_t)Ncn * HID * 4);
    float* g_comp2 = alloc((size_t)Ncn * HID * 4);
    (void)ws_size; (void)n_in;

    auto nb = [](long n) { return dim3((unsigned)((n + 255) / 256)); };
    auto gemm = [&](const float* X, const float* W, const float* bias, float* Y,
                    int N, int K, bool act) {
        dim3 g((unsigned)((N + 127) / 128));
        if (act) wmma_gemm128<1, 1><<<g, 256, 0, stream>>>(X, W, bias, Y, N, K);
        else     wmma_gemm128<0, 0><<<g, 256, 0, stream>>>(X, W, bias, Y, N, K);
    };

    // --- live-subgraph projections (ELU -> Linear -> ReLU)
    gemm(F(1), F(P1 + 2), F(P1 + 3), h_month, Nm, 32, true);
    gemm(F(2), F(P1 + 4), F(P1 + 5), h_quart, Nq, 32, true);
    gemm(F(3), F(P1 + 6), F(P1 + 7), h_year,  Ny, 32, true);
    gemm(F(4), F(P1 + 8), F(P1 + 9), h_comp,  Ncn, 32, true);

    // --- day: fused proj + TEC src-GEMM (h_day lives only in regs/LDS)
    fused_proj_tecsrc<<<dim3((unsigned)((Nd + 63) / 64)), 128, 0, stream>>>(
        F(0), F(P1 + 0), F(P1 + 1), F(TEC + 2), S_day, Nd);

    // --- TemporalEdgeConv chain (512-wide hidden folded into Wc on device)
    auto run_tec = [&](const float* Sp, const float* hsrc, int Ns, float* hdst, int Ndst,
                       const int* sI, const int* dI, const float* ea, int edim,
                       int pb, int E) {
        tec_fuse_w<<<1, 128, 0, stream>>>(F(pb), F(pb + 1), F(pb + 2), F(pb + 3), Wc, bc, edim);
        if (!Sp) { gemm(hsrc, F(pb + 2), nullptr, tmpA, Ns, 128, false); Sp = tmpA; }
        gemm(hdst, F(pb + 2) + 128 * HID, nullptr, tmpB, Ndst, 128, false);
        tec_scatter<<<nb((long)E * 32), 256, 0, stream>>>(sI, dI, ea, edim, Sp, tmpB,
                                                          Wc, bc, hdst, E);
    };
    run_tec(S_day,   nullptr, Nd, h_month, Nm, I(5),  I(6),  F(7),  4, TEC + 0,  Edm);
    run_tec(nullptr, h_month, Nm, h_quart, Nq, I(8),  I(9),  F(10), 6, TEC + 4,  Emq);
    run_tec(nullptr, h_quart, Nq, h_year,  Ny, I(11), I(12), F(13), 2, TEC + 8,  Eqy);
    run_tec(nullptr, h_year,  Ny, h_comp,  Ncn, I(14), I(15), F(16), 3, TEC + 12, Eyc);

    // --- GAT (only the live ones: gat1_qy, gat1_yc, gat2_yc)
    auto run_gat = [&](const float* hsrc, int Ns, const float* hdst, int Ndst,
                       const int* sI, const int* dI, int E, int pb, float* gout) {
        gemm(hsrc, F(pb), nullptr, tmpA, Ns, 128, false);      // z_src
        gemm(hdst, F(pb), nullptr, tmpB, Ndst, 128, false);    // z_dst
        attn_score<<<nb((long)Ns * 4), 256, 0, stream>>>(tmpA, F(pb + 1), el, Ns);
        attn_score<<<nb((long)Ndst * 4), 256, 0, stream>>>(tmpB, F(pb + 2), er, Ndst);
        fill_f32<<<nb((long)Ndst * 4), 256, 0, stream>>>(emax, -3.0e38f, Ndst * 4);
        fill_f32<<<nb((long)Ndst * 4), 256, 0, stream>>>(denom, 0.f, Ndst * 4);
        gat_init<<<nb((long)Ndst * HID), 256, 0, stream>>>(gout, hdst, F(pb + 3), Ndst);
        gat_edge_max<<<nb((long)E * 4), 256, 0, stream>>>(sI, dI, el, er, score, emax, E);
        gat_edge_exp<<<nb((long)E * 4), 256, 0, stream>>>(dI, score, emax, denom, E);
        gat_edge_scatter<<<nb((long)E * 32), 256, 0, stream>>>(sI, dI, score, denom,
                                                               tmpA, gout, E);
    };
    run_gat(h_quart, Nq, h_year, Ny,  I(11), I(12), Eqy, G1 + 8,  gbuf_y);   // gat1_qy
    run_gat(h_year,  Ny, h_comp, Ncn, I(14), I(15), Eyc, G1 + 12, gbuf_c);   // gat1_yc

    // p2 (+ outer ELU; elu(relu(x)) == relu(x))
    gemm(gbuf_y, F(P2 + 4), F(P2 + 5), g_year, Ny, 128, true);
    gemm(gbuf_c, F(P2 + 6), F(P2 + 7), g_comp, Ncn, 128, true);

    run_gat(g_year, Ny, g_comp, Ncn, I(14), I(15), Eyc, G2 + 8, g_comp2);    // gat2_yc

    readout<<<1, 128, 0, stream>>>(g_comp2, Ncn, F(79), F(80), (float*)d_out);
    (void)out_size;
}
